// WithAttention_22522808500858
// MI455X (gfx1250) — matrix operand order repaired and checked by reading, compiled, not run on hardware
//
#include <hip/hip_runtime.h>
#include <hip/hip_bf16.h>

// Problem sizes (match reference)
#define B_   64
#define S_   1024
#define H_   1024
#define A_   1024
#define I_   512
#define MTOT (B_ * S_)

// GEMM tiling for the big fused attention-logits kernel
#define BM   128
#define BN   64
#define BK   32
#define KPAD 40   // row stride in bf16 units; 80 bytes -> 16B aligned rows, bank-skewed

typedef __attribute__((ext_vector_type(16))) __bf16 bf16x16;
typedef __attribute__((ext_vector_type(2)))  __bf16 bf16x2;
typedef __attribute__((ext_vector_type(8)))  float  f32x8;

// f32 pair -> packed bf16 (hardware v_cvt_pk_bf16_f32 when available)
__device__ __forceinline__ unsigned cvt2bf(float x, float y) {
    union { bf16x2 v; unsigned u; } c;
#if __has_builtin(__builtin_amdgcn_cvt_pk_bf16_f32)
    c.v = __builtin_amdgcn_cvt_pk_bf16_f32(x, y);
#else
    c.v.x = (__bf16)x;
    c.v.y = (__bf16)y;
#endif
    return c.u;
}

__device__ __forceinline__ uint2 pack4(float4 v) {
    return make_uint2(cvt2bf(v.x, v.y), cvt2bf(v.z, v.w));
}

// Assemble a 16-element bf16 fragment from two 16B-aligned LDS chunks.
__device__ __forceinline__ bf16x16 ldsFrag16(const unsigned short* p0,
                                             const unsigned short* p1) {
    union { uint4 q[2]; bf16x16 v; } u;
    u.q[0] = *(const uint4*)p0;
    u.q[1] = *(const uint4*)p1;
    return u.v;
}

// ---------------------------------------------------------------------------
// Kernel 0: zero logits + padding flags (atomics accumulate into these)
// ---------------------------------------------------------------------------
__global__ void init_kernel(float* __restrict__ logits, unsigned* __restrict__ flags) {
    int i = blockIdx.x * blockDim.x + threadIdx.x;   // 65536 threads
    logits[i] = 0.0f;
    flags[i]  = 0u;
}

// ---------------------------------------------------------------------------
// Kernel 1: state_proj[b,a] = states[b,:]·W_dec[:,a] + b_dec[a] + b_enc[a]
// ---------------------------------------------------------------------------
__global__ void __launch_bounds__(256)
state_proj_kernel(const float* __restrict__ states, const float* __restrict__ Wdec,
                  const float* __restrict__ bdec,   const float* __restrict__ benc,
                  float* __restrict__ sp) {
    const int a = blockIdx.x * 256 + threadIdx.x;
    const int b = blockIdx.y;
    float acc = 0.0f;
    #pragma unroll 4
    for (int k = 0; k < H_; ++k)
        acc = fmaf(states[b * H_ + k], Wdec[(size_t)k * A_ + a], acc);
    sp[b * A_ + a] = acc + bdec[a] + benc[a];
}

// ---------------------------------------------------------------------------
// Kernel 2: fused  logits[b,s] += sum_a tanh(enc·W_enc + sp) * v_out[a]
//           bf16 WMMA GEMM, double-buffered LDS, fused epilogue, padding mask.
// grid = (A_/BN, MTOT/BM), block = 256 (8 waves: 4 along M, 2 along N)
// ---------------------------------------------------------------------------
__global__ void __launch_bounds__(256)
attn_logits_kernel(const float* __restrict__ enc,  const float* __restrict__ Wenc,
                   const float* __restrict__ sp,   const float* __restrict__ vout,
                   float* __restrict__ logits,     unsigned* __restrict__ flags) {
    __shared__ __align__(16) unsigned short As[2][BM * KPAD];
    __shared__ __align__(16) unsigned short Bs[2][BN * KPAD];
    __shared__ float    rowsum[BM];
    __shared__ unsigned rowflag[BM];

    const int tid   = threadIdx.x;
    const int nBase = blockIdx.x * BN;
    const int mBase = blockIdx.y * BM;
    const int lane  = tid & 31;
    const int wav   = tid >> 5;
    const int wm    = wav & 3;      // wave's M strip (32 rows)
    const int wn    = wav >> 2;     // wave's N strip (32 cols)
    const bool maskBlk = (blockIdx.x == 0);

    if (tid < BM) { rowsum[tid] = 0.0f; rowflag[tid] = 0u; }

    // ---- register staging (double buffer pipeline) ----
    float4   aS[4];
    float    bS[8];
    unsigned nzAcc[4] = {0u, 0u, 0u, 0u};
    const int bN = tid & 63;          // B-tile: this thread's n column
    const int bK = (tid >> 6) << 3;   // B-tile: 8 consecutive k values

    auto loadTiles = [&](int k0) {
        #pragma unroll
        for (int i = 0; i < 4; ++i) {               // A: 128x32 f32 as float4s
            int idx4 = tid + i * 256;
            int row  = idx4 >> 3;
            int k    = (idx4 & 7) << 2;
            const float* gp = enc + (size_t)(mBase + row) * H_ + k0 + k;
            aS[i] = *(const float4*)gp;
            if (k0 + BK < H_) __builtin_prefetch(gp + BK, 0, 1);
        }
        const float* gb = Wenc + (size_t)(k0 + bK) * A_ + nBase + bN;
        #pragma unroll
        for (int j = 0; j < 8; ++j)                 // B: coalesced across 64 n lanes
            bS[j] = gb[(size_t)j * A_];
    };
    auto stageTiles = [&](int buf) {
        #pragma unroll
        for (int i = 0; i < 4; ++i) {
            int idx4 = tid + i * 256;
            int row  = idx4 >> 3;
            int k    = (idx4 & 7) << 2;
            uint2 p = pack4(aS[i]);
            *(uint2*)&As[buf][row * KPAD + k] = p;
            if (maskBlk) nzAcc[i] |= p.x | p.y;     // bf16-packed nonzero OR
        }
        uint4 q = make_uint4(cvt2bf(bS[0], bS[1]), cvt2bf(bS[2], bS[3]),
                             cvt2bf(bS[4], bS[5]), cvt2bf(bS[6], bS[7]));
        *(uint4*)&Bs[buf][bN * KPAD + bK] = q;      // transposed: Bs[n][k], one b128
    };

    f32x8 c00 = {}, c01 = {}, c10 = {}, c11 = {};

    loadTiles(0);
    int buf = 0;
    for (int k0 = 0; k0 < H_; k0 += BK) {
        stageTiles(buf);
        __syncthreads();
        if (k0 + BK < H_) loadTiles(k0 + BK);       // overlaps with WMMAs below

        // ---- fragments (ISA 7.12.2 layouts) ----
        const int mr  = wm * 32 + (lane & 15);
        const int kbA = (lane < 16) ? 0 : 8;        // A: two 8-elem K runs per lane
        bf16x16 a0 = ldsFrag16(&As[buf][(mr     ) * KPAD + kbA], &As[buf][(mr     ) * KPAD + kbA + 16]);
        bf16x16 a1 = ldsFrag16(&As[buf][(mr + 16) * KPAD + kbA], &As[buf][(mr + 16) * KPAD + kbA + 16]);
        const int nc  = wn * 32 + (lane & 15);
        const int kbB = (lane < 16) ? 0 : 8;        // B: two 8-elem K runs per lane
        bf16x16 b0 = ldsFrag16(&Bs[buf][(nc     ) * KPAD + kbB], &Bs[buf][(nc     ) * KPAD + kbB + 16]);
        bf16x16 b1 = ldsFrag16(&Bs[buf][(nc + 16) * KPAD + kbB], &Bs[buf][(nc + 16) * KPAD + kbB + 16]);

        c00 = __builtin_amdgcn_wmma_f32_16x16x32_bf16(false, a0, false, b0, (short)0, c00, false, false);
        c01 = __builtin_amdgcn_wmma_f32_16x16x32_bf16(false, a0, false, b1, (short)0, c01, false, false);
        c10 = __builtin_amdgcn_wmma_f32_16x16x32_bf16(false, a1, false, b0, (short)0, c10, false, false);
        c11 = __builtin_amdgcn_wmma_f32_16x16x32_bf16(false, a1, false, b1, (short)0, c11, false, false);
        buf ^= 1;
    }

    // ---- fused epilogue: tanh + v_out dot, 16-lane row reduce, LDS accumulate ----
    const int bIdx = mBase / S_;   // BM divides S: one batch row per block
    #pragma unroll
    for (int ti = 0; ti < 2; ++ti) {
        #pragma unroll
        for (int tj = 0; tj < 2; ++tj) {
            const f32x8 cc = (ti == 0) ? (tj == 0 ? c00 : c01)
                                       : (tj == 0 ? c10 : c11);
            const int nl  = wn * 32 + tj * 16 + (lane & 15);
            const int ag  = nBase + nl;
            const float spv = sp[bIdx * A_ + ag];
            const float vv  = vout[ag];
            #pragma unroll
            for (int i = 0; i < 8; ++i) {
                float t = tanhf(cc[i] + spv) * vv;
                t += __shfl_xor(t, 1, 32);
                t += __shfl_xor(t, 2, 32);
                t += __shfl_xor(t, 4, 32);
                t += __shfl_xor(t, 8, 32);
                if ((lane & 15) == 0) {
                    int m = wm * 32 + ti * 16 + i + ((lane >> 4) << 3);
                    atomicAdd(&rowsum[m], t);
                }
            }
        }
    }
    if (maskBlk) {                       // padding-mask flags, once per thread
        #pragma unroll
        for (int i = 0; i < 4; ++i) {
            int row = (tid + i * 256) >> 3;
            if (nzAcc[i]) atomicOr(&rowflag[row], 1u);
        }
    }
    __syncthreads();
    if (tid < BM) {
        atomicAdd(&logits[mBase + tid], rowsum[tid]);
        if (maskBlk) flags[mBase + tid] = rowflag[tid];
    }
}

// ---------------------------------------------------------------------------
// Kernel 3: masked softmax over S per batch row; one block per b
// ---------------------------------------------------------------------------
__global__ void __launch_bounds__(256)
softmax_kernel(const float* __restrict__ logits, const unsigned* __restrict__ flags,
               const float* __restrict__ bout,   float* __restrict__ weights) {
    __shared__ float red[256];
    const int b = blockIdx.x, tid = threadIdx.x;
    float l[4], mx = -3.4e38f;
    #pragma unroll
    for (int i = 0; i < 4; ++i) {
        int s = tid + i * 256;
        float v = logits[b * S_ + s] + bout[0];
        if (flags[b * S_ + s] == 0u) v = -1e9f;
        l[i] = v;
        mx = fmaxf(mx, v);
    }
    red[tid] = mx; __syncthreads();
    for (int off = 128; off > 0; off >>= 1) {
        if (tid < off) red[tid] = fmaxf(red[tid], red[tid + off]);
        __syncthreads();
    }
    mx = red[0]; __syncthreads();
    float sum = 0.0f;
    #pragma unroll
    for (int i = 0; i < 4; ++i) { l[i] = expf(l[i] - mx); sum += l[i]; }
    red[tid] = sum; __syncthreads();
    for (int off = 128; off > 0; off >>= 1) {
        if (tid < off) red[tid] += red[tid + off];
        __syncthreads();
    }
    float inv = 1.0f / red[0];
    #pragma unroll
    for (int i = 0; i < 4; ++i)
        weights[b * S_ + tid + i * 256] = l[i] * inv;
}

// ---------------------------------------------------------------------------
// Kernel 4: attention[b,h] = sum_s weights[b,s] * encoded[b,s,h]  (BW-bound)
// one block per batch row; each thread owns 4 h columns (float4 stream)
// ---------------------------------------------------------------------------
__global__ void __launch_bounds__(256)
attn_apply_kernel(const float* __restrict__ enc, const float* __restrict__ weights,
                  float* __restrict__ attn) {
    __shared__ float w[S_];
    const int b   = blockIdx.x;
    const int tid = threadIdx.x;
    for (int i = tid; i < S_; i += 256) w[i] = weights[b * S_ + i];
    __syncthreads();
    float4 acc = make_float4(0.f, 0.f, 0.f, 0.f);
    const float* base = enc + (size_t)b * S_ * H_ + tid * 4;
    #pragma unroll 4
    for (int s = 0; s < S_; ++s) {
        float4 v = *(const float4*)(base + (size_t)s * H_);
        float ws = w[s];
        acc.x = fmaf(ws, v.x, acc.x);
        acc.y = fmaf(ws, v.y, acc.y);
        acc.z = fmaf(ws, v.z, acc.z);
        acc.w = fmaf(ws, v.w, acc.w);
    }
    *(float4*)(attn + b * H_ + tid * 4) = acc;
}

// ---------------------------------------------------------------------------
// Kernel 5: fused GRU cell with WMMA gate GEMMs.
// Block owns a 64-wide h chunk; 6 GEMMs -> 4 accumulator groups:
//   G0 = x·Wih[r] + h·Whh[r] (+biases), G1 = same for z,
//   G2 = x·Wih[n] + b_ih[n],            G3 = h·Whh[n] + b_hh[n]
// then r=sig(G0), z=sig(G1), n=tanh(G2 + r*G3), h' = (1-z)n + z h
// ---------------------------------------------------------------------------
__device__ __forceinline__ void gru_kpass(
    int srcKind /*0: x=[inputs|attn], 1: states*/, int seg, int hBase,
    const float* __restrict__ inputs, const float* __restrict__ attn,
    const float* __restrict__ states, const float* __restrict__ W, int Kd,
    unsigned short* Xs, unsigned short* Wsm, f32x8& c0, f32x8& c1, int tid) {
    const int lane = tid & 31, wav = tid >> 5;
    const int wm = wav & 3, wn = wav >> 2;
    const int jBase = seg * H_ + hBase;
    for (int k0 = 0; k0 < Kd; k0 += 32) {
        __syncthreads();
        #pragma unroll
        for (int i = 0; i < 2; ++i) {        // A tile 64x32
            int idx4 = tid + i * 256;
            int row  = idx4 >> 3;
            int k    = (idx4 & 7) << 2;
            float4 v;
            if (srcKind == 0) {
                int kg = k0 + k;
                v = (kg < I_) ? *(const float4*)(inputs + row * I_ + kg)
                              : *(const float4*)(attn + row * H_ + (kg - I_));
            } else {
                v = *(const float4*)(states + row * H_ + k0 + k);
            }
            *(uint2*)&Xs[row * KPAD + k] = pack4(v);
        }
        {   // B tile transposed: Wsm[n][k] = W[jBase+n][k0+k]; 8 consecutive k per thread, strided over n
            int n  = tid & 63;
            int kg = (tid >> 6) << 3;
            const float* gp = W + (size_t)(jBase + n) * Kd + k0 + kg;
            uint4 q = make_uint4(cvt2bf(gp[0], gp[1]), cvt2bf(gp[2], gp[3]),
                                 cvt2bf(gp[4], gp[5]), cvt2bf(gp[6], gp[7]));
            *(uint4*)&Wsm[n * KPAD + kg] = q;
        }
        __syncthreads();
        const int mr  = wm * 16 + (lane & 15);
        const int kbA = (lane < 16) ? 0 : 8;
        bf16x16 a = ldsFrag16(&Xs[mr * KPAD + kbA], &Xs[mr * KPAD + kbA + 16]);
        const int nc  = wn * 32 + (lane & 15);
        const int kbB = (lane < 16) ? 0 : 8;
        bf16x16 b0 = ldsFrag16(&Wsm[(nc     ) * KPAD + kbB], &Wsm[(nc     ) * KPAD + kbB + 16]);
        bf16x16 b1 = ldsFrag16(&Wsm[(nc + 16) * KPAD + kbB], &Wsm[(nc + 16) * KPAD + kbB + 16]);
        c0 = __builtin_amdgcn_wmma_f32_16x16x32_bf16(false, a, false, b0, (short)0, c0, false, false);
        c1 = __builtin_amdgcn_wmma_f32_16x16x32_bf16(false, a, false, b1, (short)0, c1, false, false);
    }
}

__global__ void __launch_bounds__(256)
gru_kernel(const float* __restrict__ inputs, const float* __restrict__ attn,
           const float* __restrict__ states, const float* __restrict__ Wih,
           const float* __restrict__ Whh,    const float* __restrict__ bih,
           const float* __restrict__ bhh,    float* __restrict__ out) {
    extern __shared__ __align__(16) unsigned char smem[];
    unsigned short* Xs  = (unsigned short*)smem;                    // 64*KPAD bf16
    unsigned short* Wsm = (unsigned short*)(smem + 64 * KPAD * 2);  // 64*KPAD bf16
    float* G = (float*)(smem + 2 * 64 * KPAD * 2);                  // 4 x 64x64 f32

    const int tid = threadIdx.x;
    const int hBase = blockIdx.x * 64;
    const int lane = tid & 31, wav = tid >> 5;
    const int wm = wav & 3, wn = wav >> 2;

    for (int g = 0; g < 4; ++g) {
        f32x8 c0 = {}, c1 = {};
        const int seg = (g >= 2) ? 2 : g;
        if (g != 3) gru_kpass(0, seg, hBase, inputs, attn, states, Wih, I_ + H_, Xs, Wsm, c0, c1, tid);
        if (g != 2) gru_kpass(1, seg, hBase, inputs, attn, states, Whh, H_,      Xs, Wsm, c0, c1, tid);
        float* Gg = G + g * 64 * 64;
        #pragma unroll
        for (int t = 0; t < 2; ++t) {
            const f32x8 cc = (t == 0) ? c0 : c1;
            const int n = wn * 32 + t * 16 + (lane & 15);
            float bias = 0.0f;
            if (g != 3) bias += bih[seg * H_ + hBase + n];
            if (g != 2) bias += bhh[seg * H_ + hBase + n];
            #pragma unroll
            for (int i = 0; i < 8; ++i) {
                int m = wm * 16 + i + ((lane < 16) ? 0 : 8);
                Gg[m * 64 + n] = cc[i] + bias;
            }
        }
        __syncthreads();
    }

    // gate math + output
    const float* G0 = G, *G1 = G + 4096, *G2 = G + 8192, *G3 = G + 12288;
    #pragma unroll
    for (int i = 0; i < 16; ++i) {
        int idx = tid + i * 256;            // 4096 = 64 rows x 64 cols
        int m = idx >> 6, n = idx & 63;
        float r  = 1.0f / (1.0f + expf(-G0[idx]));
        float z  = 1.0f / (1.0f + expf(-G1[idx]));
        float nn = tanhf(G2[idx] + r * G3[idx]);
        float hp = states[m * H_ + hBase + n];
        out[m * H_ + hBase + n] = (1.0f - z) * nn + z * hp;
    }
}

// ---------------------------------------------------------------------------
extern "C" void kernel_launch(void* const* d_in, const int* in_sizes, int n_in,
                              void* d_out, int out_size, void* d_ws, size_t ws_size,
                              hipStream_t stream) {
    const float* inputs = (const float*)d_in[0];
    const float* states = (const float*)d_in[1];
    const float* enc    = (const float*)d_in[2];
    const float* Wenc   = (const float*)d_in[3];
    const float* benc   = (const float*)d_in[4];
    const float* Wdec   = (const float*)d_in[5];
    const float* bdec   = (const float*)d_in[6];
    const float* vout   = (const float*)d_in[7];
    const float* bout   = (const float*)d_in[8];
    const float* Wih    = (const float*)d_in[9];
    const float* Whh    = (const float*)d_in[10];
    const float* bih    = (const float*)d_in[11];
    const float* bhh    = (const float*)d_in[12];
    float* out = (float*)d_out;

    float*    ws      = (float*)d_ws;
    float*    logits  = ws;                           // [B,S]  65536 f32
    unsigned* flags   = (unsigned*)(ws + 65536);      // [B,S]  65536 u32
    float*    weights = ws + 131072;                  // [B,S]
    float*    sp      = ws + 196608;                  // [B,A]
    float*    attn    = ws + 262144;                  // [B,H]

    init_kernel<<<256, 256, 0, stream>>>(logits, flags);
    state_proj_kernel<<<dim3(A_ / 256, B_), 256, 0, stream>>>(states, Wdec, bdec, benc, sp);
    attn_logits_kernel<<<dim3(A_ / BN, MTOT / BM), 256, 0, stream>>>(enc, Wenc, sp, vout, logits, flags);
    softmax_kernel<<<B_, 256, 0, stream>>>(logits, flags, bout, weights);
    attn_apply_kernel<<<B_, 256, 0, stream>>>(enc, weights, attn);
    size_t smemF = (size_t)2 * 64 * KPAD * 2 + (size_t)4 * 64 * 64 * 4;  // 75,776 B
    gru_kernel<<<H_ / 64, 256, smemF, stream>>>(inputs, attn, states, Wih, Whh, bih, bhh, out);
}
